// Conv1d_NN_Attn_v2_70428873719863
// MI455X (gfx1250) — matrix operand order repaired and checked by reading, not compiled, not run on hardware
//
#include <hip/hip_runtime.h>
#include <stdint.h>

typedef __attribute__((ext_vector_type(16))) __bf16 v16bf;
typedef __attribute__((ext_vector_type(8)))  float  v8f;

#define B_   8
#define C_   64
#define T_   4096
#define KNN  9
#define CK   (C_ * KNN)   // 576

__device__ __forceinline__ unsigned short f2bf(float f) {
    unsigned int u = __float_as_uint(f);
    unsigned int r = (u + 0x7FFFu + ((u >> 16) & 1u)) >> 16;  // round-to-nearest-even
    return (unsigned short)r;
}

union FragBF {
    uint4 q[2];
    v16bf v;
};

// ---------------------------------------------------------------------------
// Kernel 1: q/k/v projections; L2-normalize k,q over channels.
//   kn, qn : bf16, layout [b][t][c]  (position-major -> contiguous WMMA frags)
//   vout   : f32,  layout [b][c][t]
// ---------------------------------------------------------------------------
__global__ __launch_bounds__(256) void qkv_norm_kernel(
    const float* __restrict__ x,  const float* __restrict__ Wq,
    const float* __restrict__ Wk, const float* __restrict__ Wv,
    unsigned short* __restrict__ kn, unsigned short* __restrict__ qn,
    float* __restrict__ vout)
{
    __shared__ float Wl[3 * C_ * C_];   // 48 KB: Wq | Wk | Wv
    for (int i = threadIdx.x; i < C_ * C_; i += 256) {
        Wl[i]               = Wq[i];
        Wl[C_ * C_ + i]     = Wk[i];
        Wl[2 * C_ * C_ + i] = Wv[i];
    }
    __syncthreads();

    const int g  = blockIdx.x * 256 + threadIdx.x;
    const int bb = g >> 12;          // / 4096
    const int tt = g & (T_ - 1);

    float xv[C_];
    #pragma unroll
    for (int c = 0; c < C_; ++c) xv[c] = x[(bb * C_ + c) * T_ + tt];

    // m=0 -> k (Wk), m=1 -> q (Wq); two-pass (norm, then rescale+store) to keep regs low
    for (int m = 0; m < 2; ++m) {
        const float* wm = Wl + (m == 0 ? C_ * C_ : 0);
        unsigned short* dst = (m == 0 ? kn : qn) + (size_t)(bb * T_ + tt) * C_;
        float ss = 0.f;
        for (int d = 0; d < C_; ++d) {
            const float* wr = wm + d * C_;
            float a = 0.f;
            #pragma unroll
            for (int c = 0; c < C_; ++c) a = fmaf(wr[c], xv[c], a);
            ss = fmaf(a, a, ss);
        }
        const float inv = 1.0f / fmaxf(sqrtf(ss), 1e-12f);
        unsigned int* dst32 = (unsigned int*)dst;
        for (int d = 0; d < C_; d += 2) {
            const float* wr0 = wm + d * C_;
            const float* wr1 = wm + (d + 1) * C_;
            float a0 = 0.f, a1 = 0.f;
            #pragma unroll
            for (int c = 0; c < C_; ++c) {
                a0 = fmaf(wr0[c], xv[c], a0);
                a1 = fmaf(wr1[c], xv[c], a1);
            }
            unsigned int lo = f2bf(a0 * inv);
            unsigned int hi = f2bf(a1 * inv);
            dst32[d >> 1] = lo | (hi << 16);
        }
    }

    // v (not normalized), [b][c][t], coalesced across threads
    for (int d = 0; d < C_; ++d) {
        const float* wr = Wl + 2 * C_ * C_ + d * C_;
        float a = 0.f;
        #pragma unroll
        for (int c = 0; c < C_; ++c) a = fmaf(wr[c], xv[c], a);
        vout[(bb * C_ + d) * T_ + tt] = a;
    }
}

// ---------------------------------------------------------------------------
// Kernel 2: fused sim (WMMA bf16) -> streaming top-9 -> gather -> 1x1 conv.
// One workgroup (8 waves) owns 16 sim rows (i-positions) of one batch.
// Wave w computes j-tiles w, w+8, ... (32 tiles each, 2 WMMA per tile).
//   - B fragments double-buffered in registers; steady-state loop unrolled
//     by 2 so the buffer rotation is register renaming, not v_mov copies
//   - simT regions are wave-private: intra-wave LDS is in-order on CDNA5,
//     so only a compiler scheduling fence is needed (no block barrier)
//   - all 32 lanes run the top-9 sift: lane r scans cols 0-7 of row r,
//     lane r+16 scans cols 8-15 -> 16 partial lists merged once at the end
// ---------------------------------------------------------------------------
__global__ __launch_bounds__(256) void sim_topk_conv_kernel(
    const unsigned short* __restrict__ kn, const unsigned short* __restrict__ qn,
    const float* __restrict__ vws, const float* __restrict__ conv_w,
    const float* __restrict__ conv_b, float* __restrict__ out)
{
    __shared__ float simT[8][16][16];      // per-wave 16x16 sim tile (wave-private)
    __shared__ float topv[16][16][KNN];    // 16 partial top-9 value lists per row
    __shared__ int   topi[16][16][KNN];    // 16 partial top-9 index lists per row
    __shared__ int   fidx[16][KNN];        // merged indices, rank-descending
    __shared__ float prime[16][CK];        // gathered v, [row][c*9+kk]

    const int bb    = blockIdx.x >> 8;            // batch
    const int i0    = (blockIdx.x & 255) << 4;    // first sim row of this WG
    const int tid   = threadIdx.x;
    const int wave  = tid >> 5;
    const int lane  = tid & 31;
    const int lhalf = (lane >> 4) & 1;
    const int l16   = lane & 15;

    const unsigned short* knb = kn + (size_t)bb * T_ * C_;
    const unsigned short* qnb = qn + (size_t)bb * T_ * C_;

    // A fragments (k-rows i0..i0+15), invariant across the j sweep.
    // 16-bit A 16x32 layout (contiguous, matching B): lanes 0-15 hold K 0..15, lanes 16-31 K 16..31.
    FragBF a0, a1;
    {
        const unsigned short* ap = knb + (size_t)(i0 + l16) * C_;
        const unsigned short* p0 = ap + lhalf * 16;           // c-chunk 0..31
        a0.q[0] = *(const uint4*)(p0);
        a0.q[1] = *(const uint4*)(p0 + 8);
        const unsigned short* p1 = ap + 32 + lhalf * 16;      // c-chunk 32..63
        a1.q[0] = *(const uint4*)(p1);
        a1.q[1] = *(const uint4*)(p1 + 8);
    }

    float tv[KNN]; int ti[KNN];            // ascending: tv[0] = current min
    #pragma unroll
    for (int s = 0; s < KNN; ++s) { tv[s] = -1e30f; ti[s] = 0; }

    const int NIT = T_ / (16 * 8);         // 32 j-tiles per wave

    // B fragments (q-cols): lane holds col l16, 16 contiguous K per half.
    FragBF cb0, cb1, nb0, nb1;
    {
        const unsigned short* bp = qnb + (size_t)(wave * 16 + l16) * C_ + lhalf * 16;
        cb0.q[0] = *(const uint4*)(bp);
        cb0.q[1] = *(const uint4*)(bp + 8);
        cb1.q[0] = *(const uint4*)(bp + 32);
        cb1.q[1] = *(const uint4*)(bp + 40);
    }

    #pragma unroll 2
    for (int iter = 0; iter < NIT; ++iter) {
        const int j0 = (iter * 8 + wave) << 4;

        // prefetch next tile's B fragments (consumed next iteration)
        if (iter + 1 < NIT) {
            const int nj0 = ((iter + 1) * 8 + wave) << 4;
            const unsigned short* bp = qnb + (size_t)(nj0 + l16) * C_ + lhalf * 16;
            nb0.q[0] = *(const uint4*)(bp);
            nb0.q[1] = *(const uint4*)(bp + 8);
            nb1.q[0] = *(const uint4*)(bp + 32);
            nb1.q[1] = *(const uint4*)(bp + 40);
        }

        v8f acc = {};
        acc = __builtin_amdgcn_wmma_f32_16x16x32_bf16(false, a0.v, false, cb0.v,
                                                      (short)0, acc, false, false);
        acc = __builtin_amdgcn_wmma_f32_16x16x32_bf16(false, a1.v, false, cb1.v,
                                                      (short)0, acc, false, false);

        // C layout: VGPR r -> row r + 8*lhalf, col l16
        #pragma unroll
        for (int r = 0; r < 8; ++r)
            simT[wave][r + lhalf * 8][l16] = acc[r];

        // wave-private LDS round-trip: hardware keeps LDS in-order per wave;
        // fence only stops the compiler from reordering the ds ops.
        __builtin_amdgcn_wave_barrier();
        asm volatile("" ::: "memory");

        // all 32 lanes scan: row l16, columns lhalf*8 .. lhalf*8+7
        const float4* rp = (const float4*)&simT[wave][l16][lhalf * 8];
        const float4 v4a = rp[0];
        const float4 v4b = rp[1];
        const float cand[8] = { v4a.x, v4a.y, v4a.z, v4a.w,
                                v4b.x, v4b.y, v4b.z, v4b.w };
        #pragma unroll
        for (int cjj = 0; cjj < 8; ++cjj) {
            const float val = fmaxf(cand[cjj], 0.0f);   // clamp(sim, 0)
            const int   j   = j0 + lhalf * 8 + cjj;
            if (val > tv[0]) {                 // prune: beats current min?
                tv[0] = val; ti[0] = j;
                #pragma unroll
                for (int s = 0; s < KNN - 1; ++s) {      // branchless sift-up
                    if (tv[s] > tv[s + 1]) {
                        float fv = tv[s]; tv[s] = tv[s + 1]; tv[s + 1] = fv;
                        int   fi = ti[s]; ti[s] = ti[s + 1]; ti[s + 1] = fi;
                    }
                }
            }
        }

        __builtin_amdgcn_wave_barrier();
        asm volatile("" ::: "memory");

        cb0 = nb0; cb1 = nb1;
    }

    // publish the 16 partial lists per row (virtual wave = wave*2 + lhalf)
    {
        const int vw = wave * 2 + lhalf;
        #pragma unroll
        for (int s = 0; s < KNN; ++s) {
            topv[vw][l16][s] = tv[s];
            topi[vw][l16][s] = ti[s];
        }
    }
    __syncthreads();

    // Merge the 16 partial lists (144 candidates) -> final rank-ordered indices.
    if (tid < 16) {
        float mv[KNN]; int mi[KNN];
        #pragma unroll
        for (int s = 0; s < KNN; ++s) { mv[s] = -1e30f; mi[s] = 0; }
        for (int w = 0; w < 16; ++w) {
            for (int s = KNN - 1; s >= 0; --s) {       // descending within list
                const float val = topv[w][tid][s];
                const int   j   = topi[w][tid][s];
                if (val > mv[0]) {
                    mv[0] = val; mi[0] = j;
                    #pragma unroll
                    for (int u = 0; u < KNN - 1; ++u) {
                        if (mv[u] > mv[u + 1]) {
                            float fv = mv[u]; mv[u] = mv[u + 1]; mv[u + 1] = fv;
                            int   fi = mi[u]; mi[u] = mi[u + 1]; mi[u + 1] = fi;
                        }
                    }
                }
            }
        }
        #pragma unroll
        for (int s = 0; s < KNN; ++s) fidx[tid][s] = mi[KNN - 1 - s];  // rank 0 = best
    }
    __syncthreads();

    // Gather prime[r][c*9+kk] = v[b][c][ fidx[r][kk] ]
    const float* vb = vws + (size_t)bb * C_ * T_;
    for (int p = tid; p < 16 * CK; p += 256) {
        const int r  = p / CK;
        const int qq = p - r * CK;
        const int c  = qq / KNN;
        const int kk = qq - c * KNN;
        prime[r][qq] = vb[c * T_ + fidx[r][kk]];
    }
    __syncthreads();

    // 1x1 conv: out[b][o][i0+r] = conv_w[o,:] . prime[r,:] + conv_b[o]
    {
        const int o   = tid >> 2;
        const int sub = tid & 3;
        float a0c = conv_b[o], a1c = a0c, a2c = a0c, a3c = a0c;
        const float* wr = conv_w + (size_t)o * CK;
        for (int k = 0; k < CK; ++k) {
            const float w = wr[k];
            a0c = fmaf(w, prime[sub][k],      a0c);
            a1c = fmaf(w, prime[sub + 4][k],  a1c);
            a2c = fmaf(w, prime[sub + 8][k],  a2c);
            a3c = fmaf(w, prime[sub + 12][k], a3c);
        }
        float* ob = out + ((size_t)bb * C_ + o) * T_ + i0;
        ob[sub]      = a0c;
        ob[sub + 4]  = a1c;
        ob[sub + 8]  = a2c;
        ob[sub + 12] = a3c;
    }
}

// ---------------------------------------------------------------------------
extern "C" void kernel_launch(void* const* d_in, const int* in_sizes, int n_in,
                              void* d_out, int out_size, void* d_ws, size_t ws_size,
                              hipStream_t stream)
{
    const float* x      = (const float*)d_in[0];
    const float* Wq     = (const float*)d_in[1];
    const float* Wk     = (const float*)d_in[2];
    const float* Wv     = (const float*)d_in[3];
    const float* conv_w = (const float*)d_in[4];
    const float* conv_b = (const float*)d_in[5];
    float* out = (float*)d_out;

    // workspace: kn (4 MiB bf16) | qn (4 MiB bf16) | v (8 MiB f32)
    unsigned short* kn = (unsigned short*)d_ws;
    unsigned short* qn = kn + (size_t)B_ * T_ * C_;
    float*          vw = (float*)(qn + (size_t)B_ * T_ * C_);

    qkv_norm_kernel<<<(B_ * T_) / 256, 256, 0, stream>>>(x, Wq, Wk, Wv, kn, qn, vw);
    sim_topk_conv_kernel<<<B_ * (T_ / 16), 256, 0, stream>>>(kn, qn, vw, conv_w, conv_b, out);
}
